// MultiHeadAttentionWithVisualization_14130442404305
// MI455X (gfx1250) — matrix operand order repaired and checked by reading, not compiled, not run on hardware
//
#include <hip/hip_runtime.h>
#include <hip/hip_bf16.h>

// ---------------------------------------------------------------------------
// MultiHeadAttention on gfx1250 using V_WMMA_F32_16X16X32_F16.
// B=4, S=2048, E=1024, H=16, D=64.
// Round 2: pre-convert all fp32 operands to f16 once (memory-bound pass) so
// the GEMM inner loops are pure b128-load + wmma; 32x64 wave tiles for B reuse.
// ---------------------------------------------------------------------------

typedef __attribute__((ext_vector_type(4)))  float    v4f;
typedef __attribute__((ext_vector_type(8)))  float    v8f;
typedef __attribute__((ext_vector_type(8)))  _Float16 v8h;
typedef __attribute__((ext_vector_type(16))) _Float16 v16h;

#define S_LEN 2048
#define E_DIM 1024
#define H_NUM 16
#define D_DIM 64

__device__ __forceinline__ v16h cat8(v8h a, v8h b) {
  return __builtin_shufflevector(a, b, 0,1,2,3,4,5,6,7,8,9,10,11,12,13,14,15);
}

__device__ __forceinline__ v8f wmma16(v16h a, v16h b, v8f c) {
  // D = A(16x32 f16) * B(32x16 f16) + C(16x16 f32)
  return __builtin_amdgcn_wmma_f32_16x16x32_f16(false, a, false, b, (short)0, c,
                                                false, false);
}

// A fragment (16x32 f16, row-major source, leading dim `ld` halfs).
// Layout used here (contiguous k, same as B): lane.lo = M row; element e: K = hi*16 + e.
__device__ __forceinline__ v16h load_a_f16(const _Float16* base, int ld,
                                           int lo, int hi, int k0) {
  const _Float16* r = base + (size_t)lo * ld + k0 + hi * 16;
  return cat8(*(const v8h*)r, *(const v8h*)(r + 8));
}

// B fragment (32x16) from K-major storage Bt[N][K] (leading dim `ld` halfs).
// Layout used here (contiguous k, same as A): lane.lo = N; element e: K = hi*16 + e  -> 16 contiguous halfs.
__device__ __forceinline__ v16h load_b_f16(const _Float16* base, int ld,
                                           int lo, int hi, int k0) {
  const _Float16* r = base + (size_t)lo * ld + k0 + hi * 16;
  return cat8(*(const v8h*)r, *(const v8h*)(r + 8));
}

// ---------------------------------------------------------------------------
// Bulk fp32 -> f16 conversion (memory-bound, one pass). n must be /8.
// ---------------------------------------------------------------------------
__global__ __launch_bounds__(256) void cvt_f32_to_f16_kernel(
    const float* __restrict__ src, _Float16* __restrict__ dst, int n8)
{
  const int i = (blockIdx.x * 256 + threadIdx.x);
  if (i >= n8) return;
  const float* p = src + (size_t)i * 8;
  v4f x = *(const v4f*)p;
  v4f y = *(const v4f*)(p + 4);
  v8h r;
  r[0]=(_Float16)x[0]; r[1]=(_Float16)x[1]; r[2]=(_Float16)x[2]; r[3]=(_Float16)x[3];
  r[4]=(_Float16)y[0]; r[5]=(_Float16)y[1]; r[6]=(_Float16)y[2]; r[7]=(_Float16)y[3];
  *(v8h*)(dst + (size_t)i * 8) = r;
}

// ---------------------------------------------------------------------------
// Projection: dst = f16((X @ W^T + bias) * scale). X f16 [8192,1024],
// W f16 [1024,1024] row-major (i.e. already K-major for the B operand).
// mode 0: dst[b,h,s,d]  (Q & K layout; Q pre-scaled by D^-0.5 via `scale`)
// mode 2: dst[b,h,d,s]  (V transposed so PV GEMM's B operand is K-major)
// Block 256 = 8 waves; wave tile 32(M) x 64(N); grid (N/64, M/256).
// ---------------------------------------------------------------------------
__global__ __launch_bounds__(256) void mha_proj_kernel(
    const _Float16* __restrict__ X, const _Float16* __restrict__ W,
    const float* __restrict__ bias, _Float16* __restrict__ dst,
    int mode, float scale)
{
  const int lane = threadIdx.x & 31;
  const int wave = threadIdx.x >> 5;
  const int lo = lane & 15, hi = lane >> 4;
  const int n0 = blockIdx.x * 64;
  const int m0 = blockIdx.y * 256 + wave * 32;

  const _Float16* A0 = X + (size_t)m0 * E_DIM;
  const _Float16* A1 = A0 + (size_t)16 * E_DIM;

  v8f c[2][4];
  #pragma unroll
  for (int mt = 0; mt < 2; ++mt)
    #pragma unroll
    for (int nt = 0; nt < 4; ++nt) c[mt][nt] = (v8f)0.0f;

  for (int k0 = 0; k0 < E_DIM; k0 += 32) {
    v16h a0 = load_a_f16(A0, E_DIM, lo, hi, k0);
    v16h a1 = load_a_f16(A1, E_DIM, lo, hi, k0);
    #pragma unroll
    for (int nt = 0; nt < 4; ++nt) {
      v16h b = load_b_f16(W + (size_t)(n0 + nt * 16) * E_DIM, E_DIM, lo, hi, k0);
      c[0][nt] = wmma16(a0, b, c[0][nt]);
      c[1][nt] = wmma16(a1, b, c[1][nt]);
    }
  }

  #pragma unroll
  for (int nt = 0; nt < 4; ++nt) {
    const int n = n0 + nt * 16 + lo;
    const float bv = bias[n];
    const int h = n >> 6, d = n & 63;
    #pragma unroll
    for (int mt = 0; mt < 2; ++mt) {
      #pragma unroll
      for (int r = 0; r < 8; ++r) {
        const int m = m0 + mt * 16 + r + hi * 8;
        const int b = m >> 11, s = m & (S_LEN - 1);
        const _Float16 hv = (_Float16)((c[mt][nt][r] + bv) * scale);
        size_t idx;
        if (mode == 2)
          idx = (((size_t)(b * H_NUM + h) * D_DIM + d) * S_LEN + s);
        else
          idx = (((size_t)(b * H_NUM + h) * S_LEN + s) * D_DIM + d);
        dst[idx] = hv;
      }
    }
  }
}

// ---------------------------------------------------------------------------
// Flash attention: per wave 16 query rows x full D=64, streaming 32-key blocks.
// Q,K: f16 [B,H,S,64] (Q pre-scaled). Vt: f16 [B,H,64,S]. O: f16 [B*S, 1024].
// ---------------------------------------------------------------------------
__global__ __launch_bounds__(256) void mha_attn_kernel(
    const _Float16* __restrict__ Q, const _Float16* __restrict__ K,
    const _Float16* __restrict__ Vt, _Float16* __restrict__ O)
{
  __shared__ _Float16 Pbuf[8][16 * 32];   // per-wave P staging (C-frag -> A-frag)

  const int lane = threadIdx.x & 31;
  const int wave = threadIdx.x >> 5;
  const int lo = lane & 15, hi = lane >> 4;
  const int sb = blockIdx.x;      // 16 query blocks of 128
  const int h  = blockIdx.y;      // head
  const int b  = blockIdx.z;      // batch
  const int s0 = sb * 128 + wave * 16;

  const _Float16* Qb = Q  + ((size_t)(b * H_NUM + h) * S_LEN + s0) * D_DIM;
  const _Float16* Kb = K  + ((size_t)(b * H_NUM + h) * S_LEN) * D_DIM;
  const _Float16* Vb = Vt + ((size_t)(b * H_NUM + h) * D_DIM) * S_LEN;
  _Float16* P = Pbuf[wave];

  // Q A-fragments for the two K=32 steps over D=64 (held in regs all along).
  const v16h aq0 = load_a_f16(Qb, D_DIM, lo, hi, 0);
  const v16h aq1 = load_a_f16(Qb, D_DIM, lo, hi, 32);

  float mrow[8], lrow[8];
  #pragma unroll
  for (int r = 0; r < 8; ++r) { mrow[r] = -1e30f; lrow[r] = 0.0f; }
  v8f o[4];
  #pragma unroll
  for (int nt = 0; nt < 4; ++nt) o[nt] = (v8f)0.0f;

  for (int t0 = 0; t0 < S_LEN; t0 += 32) {
    // ---- scores S = Q @ K^T  (two 16x16 N tiles, K-dim = 64 = 2 steps) ----
    v8f sc[2];
    #pragma unroll
    for (int nt = 0; nt < 2; ++nt) {
      const _Float16* Kt = Kb + (size_t)(t0 + nt * 16) * D_DIM;
      v8f acc = (v8f)0.0f;
      acc = wmma16(aq0, load_b_f16(Kt, D_DIM, lo, hi, 0),  acc);
      acc = wmma16(aq1, load_b_f16(Kt, D_DIM, lo, hi, 32), acc);
      sc[nt] = acc;
    }

    // ---- online softmax (rows live across lanes lo=0..15 of each half) ----
    #pragma unroll
    for (int r = 0; r < 8; ++r) {
      float v = fmaxf(sc[0][r], sc[1][r]);
      v = fmaxf(v, __shfl_xor(v, 1, 32));
      v = fmaxf(v, __shfl_xor(v, 2, 32));
      v = fmaxf(v, __shfl_xor(v, 4, 32));
      v = fmaxf(v, __shfl_xor(v, 8, 32));
      const float mnew = fmaxf(mrow[r], v);
      const float corr = __expf(mrow[r] - mnew);
      mrow[r] = mnew;
      const float p0 = __expf(sc[0][r] - mnew);
      const float p1 = __expf(sc[1][r] - mnew);
      float ls = p0 + p1;
      ls += __shfl_xor(ls, 1, 32);
      ls += __shfl_xor(ls, 2, 32);
      ls += __shfl_xor(ls, 4, 32);
      ls += __shfl_xor(ls, 8, 32);
      lrow[r] = lrow[r] * corr + ls;
      #pragma unroll
      for (int nt = 0; nt < 4; ++nt) o[nt][r] *= corr;
      const int row = r + hi * 8;                  // C-frag M index
      P[row * 32 + lo]      = (_Float16)p0;        // N = lo
      P[row * 32 + 16 + lo] = (_Float16)p1;        // N = 16 + lo
    }
    asm volatile("s_wait_dscnt 0" ::: "memory");   // same-wave LDS RAW fence

    // ---- O += P @ V  (P as A-frag 16x32, Vt rows are K-major B-frags) ----
    const v16h pa = load_a_f16(P, 32, lo, hi, 0);
    #pragma unroll
    for (int nt = 0; nt < 4; ++nt) {
      const _Float16* Vr = Vb + (size_t)(nt * 16 + lo) * S_LEN + t0 + hi * 16;
      v16h bv = cat8(*(const v8h*)Vr, *(const v8h*)(Vr + 8));
      o[nt] = wmma16(pa, bv, o[nt]);
    }
  }

  // ---- epilogue: normalize, write heads-concatenated f16 [B*S, 1024] ----
  #pragma unroll
  for (int r = 0; r < 8; ++r) {
    const float inv = 1.0f / lrow[r];
    const int srow = s0 + r + hi * 8;
    #pragma unroll
    for (int nt = 0; nt < 4; ++nt) {
      const size_t idx = ((size_t)b * S_LEN + srow) * E_DIM + h * D_DIM + nt * 16 + lo;
      O[idx] = (_Float16)(o[nt][r] * inv);
    }
  }
}

// ---------------------------------------------------------------------------
// Output projection: out(fp32) = attO(f16) @ out_w16^T + out_b
// Wave tile 32(M) x 64(N); grid (N/64, M/256).
// ---------------------------------------------------------------------------
__global__ __launch_bounds__(256) void mha_outproj_kernel(
    const _Float16* __restrict__ A, const _Float16* __restrict__ W,
    const float* __restrict__ bias, float* __restrict__ out)
{
  const int lane = threadIdx.x & 31;
  const int wave = threadIdx.x >> 5;
  const int lo = lane & 15, hi = lane >> 4;
  const int n0 = blockIdx.x * 64;
  const int m0 = blockIdx.y * 256 + wave * 32;

  const _Float16* A0 = A + (size_t)m0 * E_DIM;
  const _Float16* A1 = A0 + (size_t)16 * E_DIM;

  v8f c[2][4];
  #pragma unroll
  for (int mt = 0; mt < 2; ++mt)
    #pragma unroll
    for (int nt = 0; nt < 4; ++nt) c[mt][nt] = (v8f)0.0f;

  for (int k0 = 0; k0 < E_DIM; k0 += 32) {
    v16h a0 = load_a_f16(A0, E_DIM, lo, hi, k0);
    v16h a1 = load_a_f16(A1, E_DIM, lo, hi, k0);
    #pragma unroll
    for (int nt = 0; nt < 4; ++nt) {
      v16h b = load_b_f16(W + (size_t)(n0 + nt * 16) * E_DIM, E_DIM, lo, hi, k0);
      c[0][nt] = wmma16(a0, b, c[0][nt]);
      c[1][nt] = wmma16(a1, b, c[1][nt]);
    }
  }

  #pragma unroll
  for (int nt = 0; nt < 4; ++nt) {
    const int n = n0 + nt * 16 + lo;
    const float bv = bias[n];
    #pragma unroll
    for (int mt = 0; mt < 2; ++mt) {
      #pragma unroll
      for (int r = 0; r < 8; ++r) {
        const int m = m0 + mt * 16 + r + hi * 8;
        out[(size_t)m * E_DIM + n] = c[mt][nt][r] + bv;
      }
    }
  }
}

// ---------------------------------------------------------------------------
extern "C" void kernel_launch(void* const* d_in, const int* in_sizes, int n_in,
                              void* d_out, int out_size, void* d_ws, size_t ws_size,
                              hipStream_t stream) {
  const float* query = (const float*)d_in[0];
  const float* key   = (const float*)d_in[1];
  const float* value = (const float*)d_in[2];
  const float* q_w   = (const float*)d_in[3];
  const float* q_b   = (const float*)d_in[4];
  const float* k_w   = (const float*)d_in[5];
  const float* k_b   = (const float*)d_in[6];
  const float* v_w   = (const float*)d_in[7];
  const float* v_b   = (const float*)d_in[8];
  const float* out_w = (const float*)d_in[9];
  const float* out_b = (const float*)d_in[10];
  float* out = (float*)d_out;

  // Workspace layout (f16):
  //   xq/xk/xv : activation copies   3 x 16 MiB
  //   q16/k16/vt16 : projections     3 x 16 MiB
  //   weights  : qw/kw/vw/ow         4 x  2 MiB
  //   attO     : aliases xq (dead after Q projection)
  char* ws = (char*)d_ws;
  const size_t TEN  = (size_t)8192 * 1024 * sizeof(_Float16);   // 16 MiB
  const size_t WTEN = (size_t)1024 * 1024 * sizeof(_Float16);   //  2 MiB
  _Float16* xq16 = (_Float16*)(ws + 0 * TEN);
  _Float16* xk16 = (_Float16*)(ws + 1 * TEN);
  _Float16* xv16 = (_Float16*)(ws + 2 * TEN);
  _Float16* q16  = (_Float16*)(ws + 3 * TEN);   // [B,H,S,D], pre-scaled
  _Float16* k16  = (_Float16*)(ws + 4 * TEN);   // [B,H,S,D]
  _Float16* vt16 = (_Float16*)(ws + 5 * TEN);   // [B,H,D,S] (transposed)
  _Float16* qw16 = (_Float16*)(ws + 6 * TEN + 0 * WTEN);
  _Float16* kw16 = (_Float16*)(ws + 6 * TEN + 1 * WTEN);
  _Float16* vw16 = (_Float16*)(ws + 6 * TEN + 2 * WTEN);
  _Float16* ow16 = (_Float16*)(ws + 6 * TEN + 3 * WTEN);
  _Float16* attO = xq16;                        // [B*S, E]  (buffer reuse)

  const dim3 blk(256);
  const int NX8 = (8192 * 1024) / 8;   // activation elems / 8
  const int NW8 = (1024 * 1024) / 8;   // weight elems / 8

  // 1) bulk fp32 -> f16 conversions (memory-bound)
  cvt_f32_to_f16_kernel<<<dim3(NX8 / 256), blk, 0, stream>>>(query, xq16, NX8);
  cvt_f32_to_f16_kernel<<<dim3(NX8 / 256), blk, 0, stream>>>(key,   xk16, NX8);
  cvt_f32_to_f16_kernel<<<dim3(NX8 / 256), blk, 0, stream>>>(value, xv16, NX8);
  cvt_f32_to_f16_kernel<<<dim3(NW8 / 256), blk, 0, stream>>>(q_w,   qw16, NW8);
  cvt_f32_to_f16_kernel<<<dim3(NW8 / 256), blk, 0, stream>>>(k_w,   kw16, NW8);
  cvt_f32_to_f16_kernel<<<dim3(NW8 / 256), blk, 0, stream>>>(v_w,   vw16, NW8);
  cvt_f32_to_f16_kernel<<<dim3(NW8 / 256), blk, 0, stream>>>(out_w, ow16, NW8);

  // 2) QKV projections (WMMA GEMM, 32x64 wave tiles)
  const dim3 gproj(16, 32);        // (N/64, M/256) for M=8192, N=1024
  const float scaling = 0.125f;    // HEAD_DIM^-0.5 folded into Q
  mha_proj_kernel<<<gproj, blk, 0, stream>>>(xq16, qw16, q_b, q16, 0, scaling);
  mha_proj_kernel<<<gproj, blk, 0, stream>>>(xk16, kw16, k_b, k16, 0, 1.0f);
  mha_proj_kernel<<<gproj, blk, 0, stream>>>(xv16, vw16, v_b, vt16, 2, 1.0f);

  // 3) flash attention (WMMA QK^T + PV with online softmax)
  const dim3 gattn(16, 16, 4);     // (S/128, H, B)
  mha_attn_kernel<<<gattn, blk, 0, stream>>>(q16, k16, vt16, attO);

  // 4) output projection (WMMA GEMM, fp32 result)
  mha_outproj_kernel<<<gproj, blk, 0, stream>>>(attO, ow16, out_b, out);
}
